// GRUBaseline_13022340841587
// MI455X (gfx1250) — matrix operand order repaired and checked by reading, not compiled, not run on hardware
//
#include <hip/hip_runtime.h>
#include <hip/hip_bf16.h>

typedef __attribute__((ext_vector_type(16))) _Float16 v16h;
typedef __attribute__((ext_vector_type(8)))  float    v8f;

#define HDIM  512
#define BDIM  1024
#define TDIM  512
#define PDIM  96
#define KT_HH (HDIM / 32)   // 16 K-tiles of 32 for the recurrent GEMM
#define MT    2             // row tiles per wave (M-block = 32 rows)

__device__ __forceinline__ float sigf(float x)     { return 1.0f / (1.0f + __expf(-x)); }
__device__ __forceinline__ float tanhfast(float x) { return 2.0f / (1.0f + __expf(-2.0f * x)) - 1.0f; }

// ---------------------------------------------------------------------------
// Pack W [3H x K] (f32, row-major, PyTorch gate order r,z,n) into per-lane
// WMMA B-fragments (f16). Fragment for (gate g, colTile ct, kTile kt, lane l):
//   n = l%16, kb = kt*32 + (l/16)*8, halves[i] = W[g*H + ct*16 + n][kb + i + (i>=8 ? 8 : 0)]
// stored contiguously: out[(((g*32+ct)*ktiles + kt)*32 + l)*16 + i]
// ---------------------------------------------------------------------------
__global__ void pack_weights_kernel(const float* __restrict__ W,
                                    _Float16* __restrict__ out, int ktiles) {
  int idx = blockIdx.x * blockDim.x + threadIdx.x;
  int total = 3 * 32 * ktiles * 32;
  if (idx >= total) return;
  int lane = idx & 31;
  int frag = idx >> 5;
  int kt = frag % ktiles;
  int ct = (frag / ktiles) & 31;
  int g  = frag / (ktiles * 32);
  int n  = lane & 15;
  int kb = kt * 32 + ((lane >> 4) << 3);
  int K  = ktiles * 32;
  const float* src = W + (size_t)(g * HDIM + ct * 16 + n) * K + kb;
  _Float16* dst = out + (size_t)idx * 16;
#pragma unroll
  for (int i = 0; i < 16; ++i) dst[i] = (_Float16)src[i + (i >= 8 ? 8 : 0)];
}

// bcomb[j] = bih[j] + bhh[j] for r,z blocks; bcomb[2H..3H) = bih_n only.
// bhn[j]   = bhh_n[j]  (must stay inside the r*(...) term).
__global__ void prep_bias_kernel(const float* __restrict__ bih,
                                 const float* __restrict__ bhh,
                                 float* __restrict__ bcomb,
                                 float* __restrict__ bhn) {
  int j = blockIdx.x * blockDim.x + threadIdx.x;
  if (j < 2 * HDIM)       bcomb[j] = bih[j] + bhh[j];
  else if (j < 3 * HDIM)  bcomb[j] = bih[j];
  if (j < HDIM)           bhn[j] = bhh[2 * HDIM + j];
}

__global__ void init_y_kernel(float* __restrict__ y, const float* __restrict__ dec_start) {
  int i = blockIdx.x * blockDim.x + threadIdx.x;
  if (i < BDIM) y[i] = dec_start[0];
}

// ---------------------------------------------------------------------------
// One fused GRU cell step: hout = GRUCell(xin, hin).
// Block = 128 threads (4 waves). Grid = (B/32 row-blocks, 8 col-groups).
// Each wave owns one 16-col tile of H and MT=2 row tiles (M=32), so each
// B-fragment load feeds MT WMMAs (halves L2 weight traffic vs MT=1).
// Single 32KB LDS A-buffer reused in two phases: ih GEMM, then hh GEMM.
// A-fragment ISA layout per lane l: m=l%16, kbase=kt*32+(l/16)*8,
//   halves = A[m][kbase..+7], A[m][kbase+16..+23].
// B addressing: one per-gate lane-dependent base pointer + kt*1024B immediate
// offsets -> single-VGPR-address global_load_b128 clauses.
// ---------------------------------------------------------------------------
__global__ __launch_bounds__(128) void gru_cell_kernel(
    const float* __restrict__ xin, int xstride, int kxTiles,
    const _Float16* __restrict__ WihP,   // packed ih weights (kxTiles>0)
    const float* __restrict__ Wih1,      // raw f32 [3H] ih weights (kxTiles==0)
    const float* __restrict__ hin,       // [B x H] previous state
    const _Float16* __restrict__ WhhP,   // packed hh weights (KT_HH tiles)
    const float* __restrict__ bcomb,     // [3H]
    const float* __restrict__ bhn,       // [H]
    float* __restrict__ hout)            // [B x H] new state (!= hin)
{
  __shared__ __align__(32) _Float16 ldsA[KT_HH * MT * 32 * 16];  // 32 KB

  const int rowBase = blockIdx.x * (MT * 16);
  const int lane = threadIdx.x & 31;
  const int wave = threadIdx.x >> 5;
  const int ct   = blockIdx.y * 4 + wave;   // column tile 0..31 within H

  v8f accR[MT] = {}, accZ[MT] = {}, accNi[MT] = {}, accNh[MT] = {};

  const _Float16* aB = ldsA + (size_t)lane * 16;  // lane base into LDS frags

  // ================= phase 1: ih GEMM (WMMA path) =================
  if (kxTiles > 0) {
    for (int e = threadIdx.x; e < kxTiles * MT * 32; e += blockDim.x) {
      int l = e & 31, rt = (e >> 5) & (MT - 1), kt = e >> 6;
      int m = rt * 16 + (l & 15), kb = kt * 32 + ((l >> 4) << 3);
      const float* s = xin + (size_t)(rowBase + m) * xstride + kb;
      _Float16* d = ldsA + (size_t)e * 16;
#pragma unroll
      for (int i = 0; i < 8; ++i) { d[i] = (_Float16)s[i]; d[8 + i] = (_Float16)s[16 + i]; }
    }
    __syncthreads();
    const _Float16* bR = WihP + ((size_t)(0 * 32 + ct) * kxTiles * 32 + lane) * 16;
    const _Float16* bZ = WihP + ((size_t)(1 * 32 + ct) * kxTiles * 32 + lane) * 16;
    const _Float16* bN = WihP + ((size_t)(2 * 32 + ct) * kxTiles * 32 + lane) * 16;
    for (int kt = 0; kt < kxTiles; ++kt) {
      v16h a0 = *(const v16h*)(aB + (size_t)kt * (MT * 512));
      v16h a1 = *(const v16h*)(aB + (size_t)kt * (MT * 512) + 512);
      v16h b0 = *(const v16h*)(bR + (size_t)kt * 512);
      accR[0]  = __builtin_amdgcn_wmma_f32_16x16x32_f16(false, a0, false, b0, (short)0, accR[0],  false, false);
      accR[1]  = __builtin_amdgcn_wmma_f32_16x16x32_f16(false, a1, false, b0, (short)0, accR[1],  false, false);
      v16h b1 = *(const v16h*)(bZ + (size_t)kt * 512);
      accZ[0]  = __builtin_amdgcn_wmma_f32_16x16x32_f16(false, a0, false, b1, (short)0, accZ[0],  false, false);
      accZ[1]  = __builtin_amdgcn_wmma_f32_16x16x32_f16(false, a1, false, b1, (short)0, accZ[1],  false, false);
      v16h b2 = *(const v16h*)(bN + (size_t)kt * 512);
      accNi[0] = __builtin_amdgcn_wmma_f32_16x16x32_f16(false, a0, false, b2, (short)0, accNi[0], false, false);
      accNi[1] = __builtin_amdgcn_wmma_f32_16x16x32_f16(false, a1, false, b2, (short)0, accNi[1], false, false);
    }
    __syncthreads();   // everyone done reading ldsA before re-staging
  }

  // ================= phase 2: hh GEMM =================
  for (int e = threadIdx.x; e < KT_HH * MT * 32; e += blockDim.x) {
    int l = e & 31, rt = (e >> 5) & (MT - 1), kt = e >> 6;
    int m = rt * 16 + (l & 15), kb = kt * 32 + ((l >> 4) << 3);
    const float* s = hin + (size_t)(rowBase + m) * HDIM + kb;
    _Float16* d = ldsA + (size_t)e * 16;
#pragma unroll
    for (int i = 0; i < 8; ++i) { d[i] = (_Float16)s[i]; d[8 + i] = (_Float16)s[16 + i]; }
  }
  __syncthreads();
  {
    const _Float16* bR = WhhP + ((size_t)(0 * 32 + ct) * KT_HH * 32 + lane) * 16;
    const _Float16* bZ = WhhP + ((size_t)(1 * 32 + ct) * KT_HH * 32 + lane) * 16;
    const _Float16* bN = WhhP + ((size_t)(2 * 32 + ct) * KT_HH * 32 + lane) * 16;
#pragma unroll 4
    for (int kt = 0; kt < KT_HH; ++kt) {
      v16h a0 = *(const v16h*)(aB + (size_t)kt * (MT * 512));
      v16h a1 = *(const v16h*)(aB + (size_t)kt * (MT * 512) + 512);
      v16h b0 = *(const v16h*)(bR + (size_t)kt * 512);
      accR[0]  = __builtin_amdgcn_wmma_f32_16x16x32_f16(false, a0, false, b0, (short)0, accR[0],  false, false);
      accR[1]  = __builtin_amdgcn_wmma_f32_16x16x32_f16(false, a1, false, b0, (short)0, accR[1],  false, false);
      v16h b1 = *(const v16h*)(bZ + (size_t)kt * 512);
      accZ[0]  = __builtin_amdgcn_wmma_f32_16x16x32_f16(false, a0, false, b1, (short)0, accZ[0],  false, false);
      accZ[1]  = __builtin_amdgcn_wmma_f32_16x16x32_f16(false, a1, false, b1, (short)0, accZ[1],  false, false);
      v16h b2 = *(const v16h*)(bN + (size_t)kt * 512);
      accNh[0] = __builtin_amdgcn_wmma_f32_16x16x32_f16(false, a0, false, b2, (short)0, accNh[0], false, false);
      accNh[1] = __builtin_amdgcn_wmma_f32_16x16x32_f16(false, a1, false, b2, (short)0, accNh[1], false, false);
    }
  }

  // ============ epilogue: gate fusion + state update ============
  // C/D layout: element (m = v + 8*(lane/16), n = lane%16) in VGPR v.
  const int c    = ct * 16 + (lane & 15);
  const int mOff = (lane >> 4) << 3;
  const float br  = bcomb[c];
  const float bz  = bcomb[HDIM + c];
  const float bni = bcomb[2 * HDIM + c];
  const float bnh = bhn[c];
  float wr = 0.f, wz = 0.f, wn = 0.f;
  if (kxTiles == 0) { wr = Wih1[c]; wz = Wih1[HDIM + c]; wn = Wih1[2 * HDIM + c]; }

#pragma unroll
  for (int rt = 0; rt < MT; ++rt) {
#pragma unroll
    for (int v = 0; v < 8; ++v) {
      int row = rowBase + rt * 16 + mOff + v;
      float hold = hin[(size_t)row * HDIM + c];
      float ir = accR[rt][v], iz = accZ[rt][v], ini = accNi[rt][v], inh = accNh[rt][v];
      if (kxTiles == 0) {            // rank-1 input GEMM (decoder layer 0)
        float yv = xin[row];
        ir += yv * wr; iz += yv * wz; ini += yv * wn;
      }
      float r  = sigf(ir + br);
      float z  = sigf(iz + bz);
      float nn = tanhfast(ini + bni + r * (inh + bnh));
      hout[(size_t)row * HDIM + c] = (1.0f - z) * nn + z * hold;
    }
  }
}

// y[b] = dot(d2[b,:], outW) + outb ; also scatter into d_out[b*P + p]
__global__ void proj_kernel(const float* __restrict__ d2,
                            const float* __restrict__ outW,
                            const float* __restrict__ outb,
                            float* __restrict__ y,
                            float* __restrict__ out, int p) {
  int row  = blockIdx.x * (blockDim.x >> 5) + (threadIdx.x >> 5);
  int lane = threadIdx.x & 31;
  const float* src = d2 + (size_t)row * HDIM;
  float s = 0.f;
  for (int k = lane; k < HDIM; k += 32) s += src[k] * outW[k];
#pragma unroll
  for (int off = 16; off > 0; off >>= 1) s += __shfl_down(s, off, 32);
  if (lane == 0) {
    float val = s + outb[0];
    y[row] = val;
    out[(size_t)row * PDIM + p] = val;
  }
}

extern "C" void kernel_launch(void* const* d_in, const int* in_sizes, int n_in,
                              void* d_out, int out_size, void* d_ws, size_t ws_size,
                              hipStream_t stream) {
  const float* x         = (const float*)d_in[0];
  const float* enc_Wih0  = (const float*)d_in[1];
  const float* enc_Whh0  = (const float*)d_in[2];
  const float* enc_bih0  = (const float*)d_in[3];
  const float* enc_bhh0  = (const float*)d_in[4];
  const float* enc_Wih1  = (const float*)d_in[5];
  const float* enc_Whh1  = (const float*)d_in[6];
  const float* enc_bih1  = (const float*)d_in[7];
  const float* enc_bhh1  = (const float*)d_in[8];
  const float* dec_Wih0  = (const float*)d_in[9];
  const float* dec_Whh0  = (const float*)d_in[10];
  const float* dec_bih0  = (const float*)d_in[11];
  const float* dec_bhh0  = (const float*)d_in[12];
  const float* dec_Wih1  = (const float*)d_in[13];
  const float* dec_Whh1  = (const float*)d_in[14];
  const float* dec_bih1  = (const float*)d_in[15];
  const float* dec_bhh1  = (const float*)d_in[16];
  const float* out_W     = (const float*)d_in[17];
  const float* out_b     = (const float*)d_in[18];
  const float* dec_start = (const float*)d_in[19];
  float* out = (float*)d_out;

  // Workspace bump allocator (256B aligned).
  char* ws = (char*)d_ws;
  size_t off = 0;
  auto alloc = [&](size_t bytes) -> void* {
    void* p = ws + off;
    off += (bytes + 255) & ~size_t(255);
    return p;
  };
  const size_t packBig   = sizeof(_Float16) * 3 * 32 * KT_HH * 32 * 16; // 1.5 MB
  const size_t packSmall = sizeof(_Float16) * 3 * 32 * 1     * 32 * 16; // 96 KB
  const size_t hbytes    = sizeof(float) * (size_t)BDIM * HDIM;          // 2 MB

  _Float16* encWih0P = (_Float16*)alloc(packSmall);
  _Float16* encWhh0P = (_Float16*)alloc(packBig);
  _Float16* encWih1P = (_Float16*)alloc(packBig);
  _Float16* encWhh1P = (_Float16*)alloc(packBig);
  _Float16* decWhh0P = (_Float16*)alloc(packBig);
  _Float16* decWih1P = (_Float16*)alloc(packBig);
  _Float16* decWhh1P = (_Float16*)alloc(packBig);

  float* bc_e0 = (float*)alloc(sizeof(float) * 3 * HDIM);
  float* bn_e0 = (float*)alloc(sizeof(float) * HDIM);
  float* bc_e1 = (float*)alloc(sizeof(float) * 3 * HDIM);
  float* bn_e1 = (float*)alloc(sizeof(float) * HDIM);
  float* bc_d0 = (float*)alloc(sizeof(float) * 3 * HDIM);
  float* bn_d0 = (float*)alloc(sizeof(float) * HDIM);
  float* bc_d1 = (float*)alloc(sizeof(float) * 3 * HDIM);
  float* bn_d1 = (float*)alloc(sizeof(float) * HDIM);

  float* h1a = (float*)alloc(hbytes); float* h1b = (float*)alloc(hbytes);
  float* h2a = (float*)alloc(hbytes); float* h2b = (float*)alloc(hbytes);
  float* d1a = (float*)alloc(hbytes); float* d1b = (float*)alloc(hbytes);
  float* d2a = (float*)alloc(hbytes); float* d2b = (float*)alloc(hbytes);
  float* ybuf = (float*)alloc(sizeof(float) * BDIM);

  // --- per-call prep: zero states, pack weights, fuse biases ---
  hipMemsetAsync(h1a, 0, hbytes, stream);
  hipMemsetAsync(h2a, 0, hbytes, stream);

  const int bigThreads = 3 * 32 * KT_HH * 32;   // 49152
  const int smallThreads = 3 * 32 * 1 * 32;     // 3072
  pack_weights_kernel<<<(smallThreads + 255) / 256, 256, 0, stream>>>(enc_Wih0, encWih0P, 1);
  pack_weights_kernel<<<(bigThreads + 255) / 256, 256, 0, stream>>>(enc_Whh0, encWhh0P, KT_HH);
  pack_weights_kernel<<<(bigThreads + 255) / 256, 256, 0, stream>>>(enc_Wih1, encWih1P, KT_HH);
  pack_weights_kernel<<<(bigThreads + 255) / 256, 256, 0, stream>>>(enc_Whh1, encWhh1P, KT_HH);
  pack_weights_kernel<<<(bigThreads + 255) / 256, 256, 0, stream>>>(dec_Whh0, decWhh0P, KT_HH);
  pack_weights_kernel<<<(bigThreads + 255) / 256, 256, 0, stream>>>(dec_Wih1, decWih1P, KT_HH);
  pack_weights_kernel<<<(bigThreads + 255) / 256, 256, 0, stream>>>(dec_Whh1, decWhh1P, KT_HH);

  prep_bias_kernel<<<6, 256, 0, stream>>>(enc_bih0, enc_bhh0, bc_e0, bn_e0);
  prep_bias_kernel<<<6, 256, 0, stream>>>(enc_bih1, enc_bhh1, bc_e1, bn_e1);
  prep_bias_kernel<<<6, 256, 0, stream>>>(dec_bih0, dec_bhh0, bc_d0, bn_d0);
  prep_bias_kernel<<<6, 256, 0, stream>>>(dec_bih1, dec_bhh1, bc_d1, bn_d1);
  init_y_kernel<<<4, 256, 0, stream>>>(ybuf, dec_start);

  const dim3 gGrid(BDIM / (MT * 16), 8);   // (32, 8)
  const dim3 gBlk(128);

  // --- encoder: 512 sequential steps, 2 layers each ---
  float* h1c = h1a; float* h1n = h1b;
  float* h2c = h2a; float* h2n = h2b;
  for (int t = 0; t < TDIM; ++t) {
    gru_cell_kernel<<<gGrid, gBlk, 0, stream>>>(
        x + (size_t)t * 32, TDIM * 32, 1, encWih0P, nullptr,
        h1c, encWhh0P, bc_e0, bn_e0, h1n);
    gru_cell_kernel<<<gGrid, gBlk, 0, stream>>>(
        h1n, HDIM, KT_HH, encWih1P, nullptr,
        h2c, encWhh1P, bc_e1, bn_e1, h2n);
    float* t1 = h1c; h1c = h1n; h1n = t1;
    float* t2 = h2c; h2c = h2n; h2n = t2;
  }

  // --- decoder: 96 sequential steps (d1/d2 seeded with encoder finals) ---
  float* d1c = h1c;   // final encoder layer-0 state
  float* d2c = h2c;   // final encoder layer-1 state
  for (int p = 0; p < PDIM; ++p) {
    float* d1n = (p & 1) ? d1b : d1a;
    float* d2n = (p & 1) ? d2b : d2a;
    gru_cell_kernel<<<gGrid, gBlk, 0, stream>>>(
        ybuf, 1, 0, nullptr, dec_Wih0,
        d1c, decWhh0P, bc_d0, bn_d0, d1n);
    gru_cell_kernel<<<gGrid, gBlk, 0, stream>>>(
        d1n, HDIM, KT_HH, decWih1P, nullptr,
        d2c, decWhh1P, bc_d1, bn_d1, d2n);
    proj_kernel<<<BDIM / 8, 256, 0, stream>>>(d2n, out_W, out_b, ybuf, out, p);
    d1c = d1n; d2c = d2n;
  }
}
